// TimePoseFunction_53395033424003
// MI455X (gfx1250) — compile-verified, matrix operand order checked by reading, not run on hardware
//
#include <hip/hip_runtime.h>

typedef __attribute__((ext_vector_type(16))) _Float16 v16h;
typedef __attribute__((ext_vector_type(8)))  float    v8f;

#define B_ROWS 131072
#define NCH 256
#define SLOPE_C 0.01f
#define EPS_C 1e-5f

union V16H { v16h v; uint4 q[2]; _Float16 h[16]; };
union H8   { uint4 q; _Float16 h[8]; };

// ---------------------------------------------------------------------------
// Pack fp32 weights (NCH x d_in, row-major) into f16 B-fragments laid out so
// each wave32 lane loads its 16 halves contiguously:
//   Bp[((kt*16 + ntg)*32 + lane)*16 + i]
// lane<16: N=ntg*16+lane,  K = kt*32 + {i<8 ? i : 16+(i-8)}
// lane>=16: N=ntg*16+lane-16, K = kt*32+8 + {same}
// skip_mode==1 (layer 4): padded K axis = [t/feat cols 0..32 | pad 33..63 |
//                                          activation cols 64..319 -> orig 33..288]
// ---------------------------------------------------------------------------
__global__ void pack_w_kernel(const float* __restrict__ W, _Float16* __restrict__ Bp,
                              int Kpad, int d_in, int skip_mode)
{
    int pe = blockIdx.x * 256 + threadIdx.x;
    int total = NCH * Kpad;
    if (pe >= total) return;
    int i    = pe & 15;
    int lane = (pe >> 4) & 31;
    int tile = pe >> 9;
    int ntg  = tile & 15;
    int kt   = tile >> 4;
    int n    = ntg * 16 + (lane & 15);
    int off  = (lane & 16) ? 8 : 0;
    int k    = kt * 32 + off + ((i < 8) ? i : (i + 8));
    int ko;
    if (skip_mode == 0) {
        ko = (k < d_in) ? k : -1;
    } else {
        if (k < 33)       ko = k;
        else if (k < 64)  ko = -1;
        else { ko = k - 31; if (ko >= d_in) ko = -1; }
    }
    float v = (ko >= 0) ? W[n * d_in + ko] : 0.0f;
    Bp[pe] = (_Float16)v;
}

// ---------------------------------------------------------------------------
// Hash-grid Lagrange interpolation -> padded f16 input rows [t | 32 feats | 0]
// mod is 65536 (power of two) so Python floor-mod == bitwise AND.
// ---------------------------------------------------------------------------
__global__ void interp_kernel(const float* __restrict__ t_in,
                              const float* __restrict__ F0,
                              const float* __restrict__ F1,
                              _Float16* __restrict__ x0pad)
{
    int r = blockIdx.x * 256 + threadIdx.x;
    if (r >= B_ROWS) return;
    float t  = t_in[r];
    float tm = floorf(t);
    int  tmi = (int)tm;
    float dt = t - tm;
    float lp = dt * (dt - 1.0f) * 0.5f;
    float lm = -(dt + 1.0f) * (dt - 1.0f);
    float ln = (dt + 1.0f) * dt * 0.5f;

    _Float16 row[64];
    row[0] = (_Float16)t;
    const int HASHES[2] = {73856093, 19349663};
    const float* Fs[2] = {F0, F1};
#pragma unroll
    for (int g = 0; g < 2; ++g) {
        int h  = HASHES[g];
        int idp = (((tmi - 1) ^ h) * 10) & 65535;
        int idm = (((tmi    ) ^ h) * 10) & 65535;
        int idn = (((tmi + 1) ^ h) * 10) & 65535;
        const float* fp = Fs[g] + (size_t)idp * 16;
        const float* fm = Fs[g] + (size_t)idm * 16;
        const float* fn = Fs[g] + (size_t)idn * 16;
#pragma unroll
        for (int f = 0; f < 16; ++f)
            row[1 + g * 16 + f] = (_Float16)(lp * fp[f] + lm * fm[f] + ln * fn[f]);
    }
#pragma unroll
    for (int j = 33; j < 64; ++j) row[j] = (_Float16)0.0f;

    uint4* dst = (uint4*)(x0pad + (size_t)r * 64);
    const uint4* src = (const uint4*)row;
#pragma unroll
    for (int j = 0; j < 8; ++j) dst[j] = src[j];
}

__global__ void zero_colsum_kernel(float* __restrict__ colsum)
{
    colsum[threadIdx.x] = 0.0f; // launched with 512 threads
}

// ---------------------------------------------------------------------------
// WMMA GEMM: Y[B x 256] = A @ W^T + bias (f16 in, f32 accum, f16 out).
// A = optional segment0 (x0pad, K0) ++ optional segment1 (activations, K1).
// Block = 256 threads = 8 waves laid out 4(M) x 2(N); each wave does a
// 16x128 tile via 8 v_wmma_f32_16x16x32_f16 per K-step.
// Also accumulates per-channel sum / sum-of-squares of Y for batch stats.
// ---------------------------------------------------------------------------
__global__ __launch_bounds__(256)
void gemm_bn_kernel(const _Float16* __restrict__ A0, int K0,
                    const _Float16* __restrict__ A1, int K1,
                    const _Float16* __restrict__ Bp,
                    const float* __restrict__ bias,
                    _Float16* __restrict__ Y,
                    float* __restrict__ colsum)
{
    __shared__ float    s_sum[NCH];
    __shared__ float    s_sq[NCH];
    __shared__ _Float16 s_out[8 * 16 * 128];

    int tid = threadIdx.x;
    s_sum[tid] = 0.0f;
    s_sq[tid]  = 0.0f;
    __syncthreads();

    int wave = tid >> 5;
    int lane = tid & 31;
    int mg   = wave >> 1;
    int ng   = wave & 1;
    int row_base = blockIdx.x * 64 + mg * 16;
    int n_base   = ng * 128;
    int lrow = lane & 15;
    int koff = (lane & 16) ? 8 : 0;

    v8f acc[8];
#pragma unroll
    for (int nt = 0; nt < 8; ++nt) {
        float bv = bias[n_base + nt * 16 + lrow];
#pragma unroll
        for (int r = 0; r < 8; ++r) acc[nt][r] = bv;
    }

    const _Float16* Aseg = A0;
    int Ks = K0, ktbase = 0;
    for (int seg = 0; seg < 2; ++seg) {
        if (seg == 1) { Aseg = A1; Ks = K1; ktbase = K0 >> 5; }
        if (Ks <= 0 || Aseg == nullptr) continue;
        const _Float16* arow = Aseg + (size_t)(row_base + lrow) * Ks + koff;
        int nkt = Ks >> 5;
        for (int kt = 0; kt < nkt; ++kt) {
            V16H a;
            a.q[0] = *(const uint4*)(arow + kt * 32);
            a.q[1] = *(const uint4*)(arow + kt * 32 + 16);
            const _Float16* bp = Bp + (((size_t)(ktbase + kt) * 16 + (n_base >> 4)) * 32 + lane) * 16;
#pragma unroll
            for (int nt = 0; nt < 8; ++nt) {
                V16H b;
                const uint4* bq = (const uint4*)(bp + (size_t)nt * 512);
                b.q[0] = bq[0];
                b.q[1] = bq[1];
                acc[nt] = __builtin_amdgcn_wmma_f32_16x16x32_f16(
                    false, a.v, false, b.v, (short)0, acc[nt], false, false);
            }
        }
    }

    // per-channel partial sums (8 M-rows per lane per tile)
#pragma unroll
    for (int nt = 0; nt < 8; ++nt) {
        float s = 0.0f, q = 0.0f;
#pragma unroll
        for (int r = 0; r < 8; ++r) { float v = acc[nt][r]; s += v; q += v * v; }
        int n = n_base + nt * 16 + lrow;
        atomicAdd(&s_sum[n], s);  // ds_add_f32
        atomicAdd(&s_sq[n],  q);
    }

    // stage result in LDS, then coalesced b128 global stores
    _Float16* so = &s_out[wave * 2048];
    int mhi = (lane & 16) ? 8 : 0;
#pragma unroll
    for (int nt = 0; nt < 8; ++nt)
#pragma unroll
        for (int r = 0; r < 8; ++r)
            so[(r + mhi) * 128 + nt * 16 + lrow] = (_Float16)acc[nt][r];
    __syncthreads();

    unsafeAtomicAdd(&colsum[tid],        s_sum[tid]);   // global_atomic_add_f32
    unsafeAtomicAdd(&colsum[NCH + tid],  s_sq[tid]);

    int m = lane >> 1;
    size_t grow = (size_t)(row_base + m) * NCH + n_base;
#pragma unroll
    for (int it = 0; it < 8; ++it) {
        int c = (lane & 1) + it * 2;
        *(uint4*)(Y + grow + c * 8) = *(const uint4*)(so + m * 128 + c * 8);
    }
}

__global__ void finalize_stats_kernel(const float* __restrict__ colsum,
                                      const float* __restrict__ gamma,
                                      const float* __restrict__ beta,
                                      float2* __restrict__ coeff)
{
    int n = threadIdx.x;
    float invM = 1.0f / (float)B_ROWS;
    float mu   = colsum[n] * invM;
    float var  = colsum[NCH + n] * invM - mu * mu;
    float sc   = gamma[n] * rsqrtf(var + EPS_C);
    coeff[n]   = make_float2(sc, beta[n] - mu * sc);
}

// In-place batchnorm-affine + LeakyReLU on f16 activations (8 halves/thread)
__global__ void normact_kernel(_Float16* __restrict__ Y, const float2* __restrict__ coeff)
{
    size_t idx  = (size_t)blockIdx.x * 256 + threadIdx.x;
    size_t base = idx * 8;
    int n0 = (int)(base & (NCH - 1));
    H8 d;
    d.q = *(const uint4*)(Y + base);
#pragma unroll
    for (int j = 0; j < 8; ++j) {
        float2 c = coeff[n0 + j];
        float z = c.x * (float)d.h[j] + c.y;
        d.h[j] = (_Float16)(z >= 0.0f ? z : SLOPE_C * z);
    }
    *(uint4*)(Y + base) = d.q;
}

// Heads: trans = x@Wt^T + bt ; rot = normalize(x@Wr^T + br)
__global__ __launch_bounds__(256)
void head_kernel(const _Float16* __restrict__ X,
                 const float* __restrict__ Wt, const float* __restrict__ bt,
                 const float* __restrict__ Wr, const float* __restrict__ br,
                 float* __restrict__ out)
{
    __shared__ float sw[7][NCH];
    int tid = threadIdx.x;
#pragma unroll
    for (int j = 0; j < 7; ++j)
        sw[j][tid] = (j < 3) ? Wt[j * NCH + tid] : Wr[(j - 3) * NCH + tid];
    __syncthreads();

    int r = blockIdx.x * 256 + tid;
    float acc[7];
    acc[0] = bt[0]; acc[1] = bt[1]; acc[2] = bt[2];
    acc[3] = br[0]; acc[4] = br[1]; acc[5] = br[2]; acc[6] = br[3];

    const _Float16* x = X + (size_t)r * NCH;
    for (int k = 0; k < NCH; k += 8) {
        H8 d;
        d.q = *(const uint4*)(x + k);
#pragma unroll
        for (int j = 0; j < 8; ++j) {
            float v = (float)d.h[j];
#pragma unroll
            for (int o = 0; o < 7; ++o) acc[o] += v * sw[o][k + j];
        }
    }
    out[(size_t)r * 3 + 0] = acc[0];
    out[(size_t)r * 3 + 1] = acc[1];
    out[(size_t)r * 3 + 2] = acc[2];
    float inv = rsqrtf(acc[3] * acc[3] + acc[4] * acc[4] + acc[5] * acc[5] + acc[6] * acc[6]);
    float* rot = out + (size_t)B_ROWS * 3;
    rot[(size_t)r * 4 + 0] = acc[3] * inv;
    rot[(size_t)r * 4 + 1] = acc[4] * inv;
    rot[(size_t)r * 4 + 2] = acc[5] * inv;
    rot[(size_t)r * 4 + 3] = acc[6] * inv;
}

extern "C" void kernel_launch(void* const* d_in, const int* in_sizes, int n_in,
                              void* d_out, int out_size, void* d_ws, size_t ws_size,
                              hipStream_t stream)
{
    const float *t, *F0, *F1, *Wr, *br, *Wt, *btb;
    const float *Ws[8], *bs[8], *gm[8], *be[8];

    if (in_sizes[0] == B_ROWS) {
        // dict insertion order: t, Ws[8], bs[8], gammas[8], betas[8], F[2], Wr, br, Wt, bt
        t = (const float*)d_in[0];
        for (int i = 0; i < 8; ++i) Ws[i] = (const float*)d_in[1 + i];
        for (int i = 0; i < 8; ++i) bs[i] = (const float*)d_in[9 + i];
        for (int i = 0; i < 8; ++i) gm[i] = (const float*)d_in[17 + i];
        for (int i = 0; i < 8; ++i) be[i] = (const float*)d_in[25 + i];
        F0 = (const float*)d_in[33]; F1 = (const float*)d_in[34];
        Wr = (const float*)d_in[35]; br = (const float*)d_in[36];
        Wt = (const float*)d_in[37]; btb = (const float*)d_in[38];
    } else {
        // jax tree_flatten sorted-key order:
        // F0,F1, Wr, Ws0-7, Wt, betas0-7, br, bs0-7, bt, gammas0-7, t
        F0 = (const float*)d_in[0]; F1 = (const float*)d_in[1];
        Wr = (const float*)d_in[2];
        for (int i = 0; i < 8; ++i) Ws[i] = (const float*)d_in[3 + i];
        Wt = (const float*)d_in[11];
        for (int i = 0; i < 8; ++i) be[i] = (const float*)d_in[12 + i];
        br = (const float*)d_in[20];
        for (int i = 0; i < 8; ++i) bs[i] = (const float*)d_in[21 + i];
        btb = (const float*)d_in[29];
        for (int i = 0; i < 8; ++i) gm[i] = (const float*)d_in[30 + i];
        t = (const float*)d_in[38];
    }

    char* wsb = (char*)d_ws;
    size_t off = 0;
    auto alloc = [&](size_t bytes) {
        void* p = wsb + off;
        off += (bytes + 255) & ~(size_t)255;
        return p;
    };
    _Float16* x0pad = (_Float16*)alloc((size_t)B_ROWS * 64 * 2);
    _Float16* Ya    = (_Float16*)alloc((size_t)B_ROWS * NCH * 2);
    _Float16* Yb    = (_Float16*)alloc((size_t)B_ROWS * NCH * 2);
    _Float16* Bp[8];
    for (int i = 0; i < 8; ++i) Bp[i] = (_Float16*)alloc((size_t)NCH * 320 * 2);
    float*  colsum = (float*)alloc(512 * sizeof(float));
    float2* coeff  = (float2*)alloc(NCH * sizeof(float2));

    const int Kpad[8] = {64, 256, 256, 256, 320, 256, 256, 256};
    const int din[8]  = {33, 256, 256, 256, 289, 256, 256, 256};
    for (int i = 0; i < 8; ++i) {
        int total = NCH * Kpad[i];
        pack_w_kernel<<<(total + 255) / 256, 256, 0, stream>>>(
            Ws[i], Bp[i], Kpad[i], din[i], (i == 4) ? 1 : 0);
    }

    interp_kernel<<<B_ROWS / 256, 256, 0, stream>>>(t, F0, F1, x0pad);

    _Float16* bufs[2] = {Ya, Yb};
    _Float16* Xcur = nullptr;
    for (int L = 0; L < 8; ++L) {
        _Float16* Yout = bufs[L & 1];
        zero_colsum_kernel<<<1, 512, 0, stream>>>(colsum);
        if (L == 0)
            gemm_bn_kernel<<<B_ROWS / 64, 256, 0, stream>>>(
                x0pad, 64, nullptr, 0, Bp[0], bs[0], Yout, colsum);
        else if (L == 4)
            gemm_bn_kernel<<<B_ROWS / 64, 256, 0, stream>>>(
                x0pad, 64, Xcur, 256, Bp[4], bs[4], Yout, colsum);
        else
            gemm_bn_kernel<<<B_ROWS / 64, 256, 0, stream>>>(
                Xcur, 256, nullptr, 0, Bp[L], bs[L], Yout, colsum);
        finalize_stats_kernel<<<1, 256, 0, stream>>>(colsum, gm[L], be[L], coeff);
        normact_kernel<<<(B_ROWS * NCH / 8) / 256, 256, 0, stream>>>(Yout, coeff);
        Xcur = Yout;
    }

    head_kernel<<<B_ROWS / 256, 256, 0, stream>>>(Xcur, Wt, btb, Wr, br, (float*)d_out);
}
